// SelfAttention_5291399708719
// MI455X (gfx1250) — matrix operand order repaired and checked by reading, not compiled, not run on hardware
//
#include <hip/hip_runtime.h>
#include <hip/hip_bf16.h>
#include <cmath>

typedef _Float16 h16;
typedef __attribute__((ext_vector_type(8)))  _Float16 v8h;
typedef __attribute__((ext_vector_type(16))) _Float16 v16h;
typedef __attribute__((ext_vector_type(8)))  float    v8f;

#define CHN  256
#define NBAT 8
#define NPIX 4096   // 64*64
#define KF   32     // CH/8
#define KH   128    // CH/2

static __device__ __forceinline__ v16h cat8(v8h lo, v8h hi) {
  return __builtin_shufflevector(lo, hi, 0,1,2,3,4,5,6,7,8,9,10,11,12,13,14,15);
}
static __device__ __forceinline__ v8f zero8f() {
  v8f z = {0.f,0.f,0.f,0.f,0.f,0.f,0.f,0.f};
  return z;
}
static __device__ __forceinline__ v8f wmma_f16(v16h a, v16h b, v8f c) {
  // D = A(16x32 f16) * B(32x16 f16) + C(16x16 f32)
  return __builtin_amdgcn_wmma_f32_16x16x32_f16(false, a, false, b, (short)0, c, false, false);
}

// ---------------------------------------------------------------------------
// K1: projections. Produces:
//   Ft[b][n][32]  f16 (transposed f)   Gt[b][n][32] f16 (transposed g)
//   Hm[b][128][4096] f16 (row-major h)
// Block: 256 threads, handles 32 n-positions of one batch; x staged via LDS.
// ---------------------------------------------------------------------------
__global__ void __launch_bounds__(256)
proj_kernel(const float* __restrict__ x,
            const float* __restrict__ wf, const float* __restrict__ bf,
            const float* __restrict__ wg, const float* __restrict__ bg,
            const float* __restrict__ wh, const float* __restrict__ bh,
            h16* __restrict__ Ft, h16* __restrict__ Gt, h16* __restrict__ Hm) {
  __shared__ float xs[32][33];
  const int t  = threadIdx.x;
  const int b  = blockIdx.x >> 7;           // / 128
  const int n0 = (blockIdx.x & 127) << 5;   // * 32
  const int nl = t & 31;
  const int ks = t >> 5;                    // 0..7

  float acc[24];
#pragma unroll
  for (int j = 0; j < 24; ++j) acc[j] = 0.f;

  for (int cc = 0; cc < 8; ++cc) {
    __syncthreads();
#pragma unroll
    for (int i = 0; i < 4; ++i) {
      int cl = ks + (i << 3);
      xs[cl][nl] = x[(((size_t)b * CHN + cc * 32 + cl) * NPIX) + n0 + nl];
    }
    __syncthreads();
#pragma unroll
    for (int j = 0; j < 24; ++j) {
      int kg = ks + (j << 3);                // 0..191 stacked: f(0-31) g(32-63) h(64-191)
      const float* wr = (kg < 32) ? (wf + kg * CHN)
                       : (kg < 64) ? (wg + (kg - 32) * CHN)
                                   : (wh + (kg - 64) * CHN);
      wr += cc * 32;
      float s = acc[j];
#pragma unroll
      for (int ci = 0; ci < 32; ++ci) s = fmaf(wr[ci], xs[ci][nl], s);
      acc[j] = s;
    }
  }

  const int n = n0 + nl;
#pragma unroll
  for (int j = 0; j < 24; ++j) {
    int kg = ks + (j << 3);
    if (kg < 32) {
      Ft[((size_t)b * NPIX + n) * KF + kg] = (h16)(acc[j] + bf[kg]);
    } else if (kg < 64) {
      Gt[((size_t)b * NPIX + n) * KF + (kg - 32)] = (h16)(acc[j] + bg[kg - 32]);
    } else {
      Hm[((size_t)b * KH + (kg - 64)) * NPIX + n] = (h16)(acc[j] + bh[kg - 64]);
    }
  }
}

// ---------------------------------------------------------------------------
// K2: per-row softmax stats. s[n][m] = sum_k Ft[n][k]*Gt[m][k].
// Two passes to keep the inner loops WMMA-bound:
//   pass 1: row max only (8 v_max per tile)
//   pass 2: recompute scores, sum exp(s - rowmax) (8 exp + 8 add per tile)
// B operand is software-pipelined one tile ahead.
// ---------------------------------------------------------------------------
__global__ void __launch_bounds__(32)
rowstats_kernel(const h16* __restrict__ Ft, const h16* __restrict__ Gt,
                float* __restrict__ rowmax, float* __restrict__ rowinv) {
  const int lane = threadIdx.x;
  const int b  = blockIdx.x >> 8;
  const int n0 = (blockIdx.x & 255) << 4;
  const bool low = lane < 16;
  const int lr   = lane & 15;
  const int off0 = low ? 0 : 8;

  // A operand: Ft rows n0..n0+15 (contiguous k order: lane halves k {0..15} / {16..31})
  const h16* fr = Ft + ((size_t)b * NPIX + n0 + lr) * KF;
  const v16h a = cat8(*(const v8h*)(fr + 2 * off0), *(const v8h*)(fr + 2 * off0 + 8));

  const h16* gbase = Gt + ((size_t)b * NPIX + lr) * KF + (low ? 0 : 16);

  float rmax[8];
#pragma unroll
  for (int r = 0; r < 8; ++r) rmax[r] = -3.0e38f;

  // ---- pass 1: max ----
  v16h bm = *(const v16h*)gbase;
  for (int mt = 0; mt < 256; ++mt) {
    v16h bnext = (mt < 255) ? *(const v16h*)(gbase + (size_t)(mt + 1) * 16 * KF) : bm;
    v8f s = wmma_f16(a, bm, zero8f());
#pragma unroll
    for (int r = 0; r < 8; ++r) rmax[r] = fmaxf(rmax[r], s[r]);
    bm = bnext;
  }
  // reduce 16 m-lanes per row (xor<16 stays within each half)
#pragma unroll
  for (int d = 1; d < 16; d <<= 1) {
#pragma unroll
    for (int r = 0; r < 8; ++r) rmax[r] = fmaxf(rmax[r], __shfl_xor(rmax[r], d, 32));
  }

  // ---- pass 2: sum of exp(s - rowmax) ----
  float rsum[8];
#pragma unroll
  for (int r = 0; r < 8; ++r) rsum[r] = 0.f;

  bm = *(const v16h*)gbase;
  for (int mt = 0; mt < 256; ++mt) {
    v16h bnext = (mt < 255) ? *(const v16h*)(gbase + (size_t)(mt + 1) * 16 * KF) : bm;
    v8f s = wmma_f16(a, bm, zero8f());
#pragma unroll
    for (int r = 0; r < 8; ++r) rsum[r] += __expf(s[r] - rmax[r]);
    bm = bnext;
  }
#pragma unroll
  for (int d = 1; d < 16; d <<= 1) {
#pragma unroll
    for (int r = 0; r < 8; ++r) rsum[r] += __shfl_xor(rsum[r], d, 32);
  }

  if (lr == 0) {
#pragma unroll
    for (int r = 0; r < 8; ++r) {
      int n = n0 + r + (low ? 0 : 8);
      rowmax[(size_t)b * NPIX + n] = rmax[r];
      rowinv[(size_t)b * NPIX + n] = 1.0f / rsum[r];
    }
  }
}

// ---------------------------------------------------------------------------
// K3: fused flash-style apply + 1x1 v-conv + residual.
// One wave per 32-m tile (two 16-col subtiles -> 2x Hm reuse).
// Loop n in chunks of 32: prefetch Hm A-tiles, score WMMAs -> exp/scale ->
// P^T via LDS -> 16 apply WMMAs into o[128x32] regs; then out = x+g*(wv*o+bv).
// ---------------------------------------------------------------------------
__global__ void __launch_bounds__(32)
attn_kernel(const h16* __restrict__ Ft, const h16* __restrict__ Gt,
            const h16* __restrict__ Hm,
            const float* __restrict__ rowmax, const float* __restrict__ rowinv,
            const float* __restrict__ x, const float* __restrict__ wv,
            const float* __restrict__ bv, const float* __restrict__ gamma,
            float* __restrict__ out) {
  __shared__ alignas(16) h16 Pt[32 * 32];    // P^T: [m 0..31][n_local 0..31]
  __shared__ alignas(16) h16 Ot[32 * 128];   // o^T: [m 0..31][c 0..127]

  const int lane = threadIdx.x;
  const int b  = blockIdx.x >> 7;            // / 128
  const int m0 = (blockIdx.x & 127) << 5;    // * 32
  const bool low = lane < 16;
  const int lr   = lane & 15;
  const int off0 = low ? 0 : 8;

  // fixed score B operands: Gt tiles at m0, m0+16
  v16h bscore[2];
#pragma unroll
  for (int mh = 0; mh < 2; ++mh) {
    const h16* gr = Gt + ((size_t)b * NPIX + m0 + mh * 16 + lr) * KF + (low ? 0 : 16);
    bscore[mh] = *(const v16h*)gr;
  }

  const float* rm = rowmax + (size_t)b * NPIX;
  const float* ri = rowinv + (size_t)b * NPIX;

  v8f oacc[2][8];
#pragma unroll
  for (int mh = 0; mh < 2; ++mh)
#pragma unroll
    for (int ct = 0; ct < 8; ++ct) oacc[mh][ct] = zero8f();

  for (int nb = 0; nb < NPIX; nb += 32) {
    // prefetch the 8 Hm A-tiles for this n-chunk (hides latency behind exp/LDS)
    v16h ah[8];
#pragma unroll
    for (int ct = 0; ct < 8; ++ct) {
      const h16* hr = Hm + ((size_t)b * KH + ct * 16 + lr) * NPIX + nb;
      ah[ct] = cat8(*(const v8h*)(hr + 2 * off0), *(const v8h*)(hr + 2 * off0 + 8));
    }

    // scores -> P^T in LDS
#pragma unroll
    for (int t = 0; t < 2; ++t) {
      const h16* fr = Ft + ((size_t)b * NPIX + nb + t * 16 + lr) * KF;
      v16h a = cat8(*(const v8h*)(fr + 2 * off0), *(const v8h*)(fr + 2 * off0 + 8));
      // softmax row stats depend on n only -> load once, reuse for both m-halves
      float mrow[8], irow[8];
#pragma unroll
      for (int r = 0; r < 8; ++r) {
        int n = nb + t * 16 + r + (low ? 0 : 8);
        mrow[r] = rm[n];
        irow[r] = ri[n];
      }
#pragma unroll
      for (int mh = 0; mh < 2; ++mh) {
        v8f s = wmma_f16(a, bscore[mh], zero8f());
        v8h ph;
#pragma unroll
        for (int r = 0; r < 8; ++r) ph[r] = (h16)(__expf(s[r] - mrow[r]) * irow[r]);
        // lane holds 8 consecutive-n values of column m -> one 16B DS store
        *(v8h*)&Pt[(mh * 16 + lr) * 32 + t * 16 + off0] = ph;
      }
    }

    // re-read P^T as B operands (lane = column m, 16 consecutive K=n)
    v16h bp[2];
#pragma unroll
    for (int mh = 0; mh < 2; ++mh)
      bp[mh] = *(const v16h*)(Pt + (mh * 16 + lr) * 32 + (low ? 0 : 16));

    // apply: 8 c-tiles x 2 m-halves, A reused across m-halves
#pragma unroll
    for (int ct = 0; ct < 8; ++ct) {
#pragma unroll
      for (int mh = 0; mh < 2; ++mh)
        oacc[mh][ct] = wmma_f16(ah[ct], bp[mh], oacc[mh][ct]);
    }
  }

  // stage o^T (f16) to LDS for the v-conv B operand
#pragma unroll
  for (int mh = 0; mh < 2; ++mh) {
#pragma unroll
    for (int ct = 0; ct < 8; ++ct) {
      v8h oh;
#pragma unroll
      for (int r = 0; r < 8; ++r) oh[r] = (h16)oacc[mh][ct][r];
      *(v8h*)&Ot[(mh * 16 + lr) * 128 + ct * 16 + off0] = oh;
    }
  }

  const float g = gamma[0];
  for (int mt = 0; mt < 16; ++mt) {          // 16 cout tiles of 16
    v8f acc[2] = {zero8f(), zero8f()};
#pragma unroll
    for (int ks = 0; ks < 4; ++ks) {         // K = 128 = 4 x 32
      const float* wr = wv + (size_t)(mt * 16 + lr) * KH + ks * 32 + 2 * off0;
      v8h wlo, whi;
#pragma unroll
      for (int i = 0; i < 8; ++i) { wlo[i] = (h16)wr[i]; whi[i] = (h16)wr[8 + i]; }
      v16h aw = cat8(wlo, whi);
#pragma unroll
      for (int mh = 0; mh < 2; ++mh) {
        v16h bo = *(const v16h*)(Ot + (mh * 16 + lr) * 128 + ks * 32 + (low ? 0 : 16));
        acc[mh] = wmma_f16(aw, bo, acc[mh]);
      }
    }
#pragma unroll
    for (int mh = 0; mh < 2; ++mh) {
#pragma unroll
      for (int r = 0; r < 8; ++r) {
        int cout   = mt * 16 + r + (low ? 0 : 8);
        size_t idx = (((size_t)b * CHN + cout) * NPIX) + m0 + mh * 16 + lr;
        out[idx] = x[idx] + g * (acc[mh][r] + bv[cout]);
      }
    }
  }
}

// ---------------------------------------------------------------------------
extern "C" void kernel_launch(void* const* d_in, const int* in_sizes, int n_in,
                              void* d_out, int out_size, void* d_ws, size_t ws_size,
                              hipStream_t stream) {
  (void)in_sizes; (void)n_in; (void)out_size; (void)ws_size;
  const float* x     = (const float*)d_in[0];
  const float* wf    = (const float*)d_in[1];
  const float* bf    = (const float*)d_in[2];
  const float* wg    = (const float*)d_in[3];
  const float* bg    = (const float*)d_in[4];
  const float* wh    = (const float*)d_in[5];
  const float* bh    = (const float*)d_in[6];
  const float* wv    = (const float*)d_in[7];
  const float* bv    = (const float*)d_in[8];
  const float* gamma = (const float*)d_in[9];
  float* out = (float*)d_out;

  char* ws = (char*)d_ws;
  size_t off = 0;
  auto carve = [&](size_t bytes) -> void* {
    off = (off + 255) & ~(size_t)255;
    void* p = ws + off;
    off += bytes;
    return p;
  };
  h16*   Ft     = (h16*)  carve((size_t)NBAT * NPIX * KF * sizeof(h16));   // 2 MB
  h16*   Gt     = (h16*)  carve((size_t)NBAT * NPIX * KF * sizeof(h16));   // 2 MB
  h16*   Hm     = (h16*)  carve((size_t)NBAT * KH * NPIX * sizeof(h16));   // 8 MB
  float* rowmax = (float*)carve((size_t)NBAT * NPIX * sizeof(float));      // 128 KB
  float* rowinv = (float*)carve((size_t)NBAT * NPIX * sizeof(float));      // 128 KB

  proj_kernel<<<NBAT * (NPIX / 32), 256, 0, stream>>>(
      x, wf, bf, wg, bg, wh, bh, Ft, Gt, Hm);
  rowstats_kernel<<<NBAT * (NPIX / 16), 32, 0, stream>>>(
      Ft, Gt, rowmax, rowinv);
  attn_kernel<<<NBAT * (NPIX / 32), 32, 0, stream>>>(
      Ft, Gt, Hm, rowmax, rowinv, x, wv, bv, gamma, out);
}
